// Transformer_16982300689118
// MI455X (gfx1250) — matrix operand order repaired and checked by reading, not compiled, not run on hardware
//
#include <hip/hip_runtime.h>

// ---------------- Model constants (match reference) ----------------
#define Bx 2
#define Sx 2048
#define DIMx 512
#define Hx 8
#define FFx 2048
#define Lx 6
#define Vx 32000
#define HDx 64            // DIM / H
#define EPSx 1e-5f
#define NEGx (-1000000000.0f)
#define Mx (Bx * Sx)      // 4096 rows

// ---------------- WMMA / TDM vector types ----------------
typedef __attribute__((ext_vector_type(16))) __bf16 v16bf;
typedef __attribute__((ext_vector_type(8)))  float  v8f;
typedef __attribute__((ext_vector_type(4)))  unsigned u32x4;
typedef __attribute__((ext_vector_type(8)))  int      i32x8;
typedef __attribute__((ext_vector_type(4)))  int      i32x4;

// Native bf16 convert (lowered to v_cvt_pk_bf16_f32 when pairable)
__device__ __forceinline__ __bf16 f2bf(float f) { return (__bf16)f; }

__device__ __forceinline__ v8f wmma_bf16(v16bf a, v16bf b, v8f c) {
  return __builtin_amdgcn_wmma_f32_16x16x32_bf16(false, a, false, b, (short)0, c, false, false);
}

// ---- bf16 fragment loads (fragment element order == memory order) ----
// A-frag: v[0..15] = p[kb..kb+15]   (16 contiguous)
__device__ __forceinline__ v16bf load_afrag(const __bf16* rowp, int kb) {
  union { v16bf v; uint4 q[2]; } u;
  u.q[0] = *reinterpret_cast<const uint4*>(rowp + kb);
  u.q[1] = *reinterpret_cast<const uint4*>(rowp + kb + 8);
  return u.v;
}
// B-frag: v[0..15] = p[0..15]  (16 contiguous, 32B aligned)
__device__ __forceinline__ v16bf load_bfrag(const __bf16* p) {
  union { v16bf v; uint4 q[2]; } u;
  u.q[0] = *reinterpret_cast<const uint4*>(p);
  u.q[1] = *reinterpret_cast<const uint4*>(p + 8);
  return u.v;
}

// ---------------- permlane16 butterfly reductions (16-lane halves) --------
__device__ __forceinline__ float plane16(float v, unsigned lo, unsigned hi) {
  unsigned s = (unsigned)__float_as_int(v);
  unsigned r = __builtin_amdgcn_permlane16(s, s, lo, hi, false, false);
  return __int_as_float((int)r);
}
__device__ __forceinline__ float redmax16(float v) {
  v = fmaxf(v, plane16(v, 0x67452301u, 0xEFCDAB89u));  // xor 1
  v = fmaxf(v, plane16(v, 0x54761032u, 0xDCFE98BAu));  // xor 2
  v = fmaxf(v, plane16(v, 0x32107654u, 0xBA98FEDCu));  // xor 4
  v = fmaxf(v, plane16(v, 0xFEDCBA98u, 0x76543210u));  // xor 8
  return v;
}
__device__ __forceinline__ float redsum16(float v) {
  v += plane16(v, 0x67452301u, 0xEFCDAB89u);
  v += plane16(v, 0x54761032u, 0xDCFE98BAu);
  v += plane16(v, 0x32107654u, 0xBA98FEDCu);
  v += plane16(v, 0xFEDCBA98u, 0x76543210u);
  return v;
}

// ---------------- Tensor Data Mover: 2D tile -> LDS ----------------
// D# per CDNA5 ISA. tensor dims == tile dims (no OOB),
// stride = matrix row stride. Optional LDS pad: 1 DWORD per 32 DWORDs.
// 6-arg builtin (clang-23 / therock-10.0 toolchain).
__device__ __forceinline__ void tdm_load_2d(void* lds_ptr, const float* gptr,
                                            unsigned tile_w, unsigned tile_h,
                                            unsigned stride, bool pad) {
  unsigned long long ga = (unsigned long long)(size_t)gptr;
  u32x4 g0;
  g0[0] = 1u;                                            // count=1
  g0[1] = (unsigned)(size_t)lds_ptr;                     // lds byte address
  g0[2] = (unsigned)ga;                                  // global_addr[31:0]
  g0[3] = ((unsigned)(ga >> 32) & 0x1ffffffu) | (2u << 30);  // ga[56:32] | type=2
  unsigned w0 = (2u << 16);                              // data_size=2 (4B)
  if (pad) w0 |= (1u << 20) | (4u << 22);                // pad 1 DW / 32 DW
  i32x8 g1;
  g1[0] = (int)w0;
  g1[1] = (int)(tile_w << 16);                           // tensor_dim0[15:0]
  g1[2] = (int)(tile_h << 16);                           // tensor_dim1[15:0]
  g1[3] = (int)(tile_w << 16);                           // tile_dim0
  g1[4] = (int)tile_h;                                   // tile_dim1
  g1[5] = (int)stride;                                   // tensor_dim0_stride
  g1[6] = 0;
  g1[7] = 0;
  i32x4 z4 = {0, 0, 0, 0};
  i32x8 z8 = {0, 0, 0, 0, 0, 0, 0, 0};
  __builtin_amdgcn_tensor_load_to_lds(g0, g1, z4, z4, z8, 0);
}

// =====================================================================
// GEMM: C[M,N] = A[M,K] f32 x W[K,N] f32 + bias [+residual] [ReLU]
// Output mode OM: 0 = f32, 1 = bf16, 2 = bf16 transposed (C^T[n][row],
// used for V so attention's P*V B-fragments become contiguous loads).
// Block 256 threads = 8 waves; block tile 64x128; wave = 1 A-frag x 4 WMMAs.
// Tiles DMA'd to LDS by the TDM, double-buffered (TENSORcnt pipelining).
// =====================================================================
template <bool RELU, bool RES, int OM>
__global__ __launch_bounds__(256) void gemm_kernel(
    const float* __restrict__ A, const float* __restrict__ W,
    const float* __restrict__ bias, const float* __restrict__ Rsd,
    void* __restrict__ Cv, int M, int N, int K) {
  __shared__ float Asf[2][64 * 33];    // 64 rows x 32 k + 1 DW TDM row pad
  __shared__ float Wsf[2][32][128];    // [k][n]
  const int t = threadIdx.x;
  const int lane = t & 31, wave = t >> 5;
  const int wrow = wave >> 1, wcol = wave & 1;   // 4 x 2 wave grid
  const int col = lane & 15, half = lane >> 4;
  const int m0 = blockIdx.y * 64, n0 = blockIdx.x * 128;

  v8f acc[4];
  {
    v8f z = {0.f, 0.f, 0.f, 0.f, 0.f, 0.f, 0.f, 0.f};
#pragma unroll
    for (int s = 0; s < 4; ++s) acc[s] = z;
  }

  const int T = K >> 5;
  if (wave == 0) {
    tdm_load_2d(&Asf[0][0], A + (size_t)m0 * K, 32, 64, (unsigned)K, true);
    tdm_load_2d(&Wsf[0][0][0], W + n0, 128, 32, (unsigned)N, false);
  }
  for (int tt = 0; tt < T; ++tt) {
    if (wave == 0) {
      if (tt + 1 < T) {
        int k1 = (tt + 1) << 5;
        tdm_load_2d(&Asf[(tt + 1) & 1][0], A + (size_t)m0 * K + k1,
                    32, 64, (unsigned)K, true);
        tdm_load_2d(&Wsf[(tt + 1) & 1][0][0], W + (size_t)k1 * N + n0,
                    128, 32, (unsigned)N, false);
        __builtin_amdgcn_s_wait_tensorcnt(2);   // current tile's 2 ops done
      } else {
        __builtin_amdgcn_s_wait_tensorcnt(0);
      }
    }
    __syncthreads();
    const int bufi = tt & 1;
    const float* Ab = &Asf[bufi][0];
    // A fragment (16x32): lane(0-15) row=lane; k in {kb..kb+7, 16+kb..+7}
    v16bf a;
    {
      int arow = wrow * 16 + col, kb = half * 8;
#pragma unroll
      for (int i = 0; i < 8; ++i) {
        a[i]     = f2bf(Ab[arow * 33 + kb + i]);
        a[8 + i] = f2bf(Ab[arow * 33 + 16 + kb + i]);
      }
    }
#pragma unroll
    for (int sub = 0; sub < 4; ++sub) {
      v16bf bfr;
      int bn = wcol * 64 + sub * 16 + col;
#pragma unroll
      for (int i = 0; i < 16; ++i) bfr[i] = f2bf(Wsf[bufi][(i < 8 ? half * 8 + i : 8 + half * 8 + i)][bn]);
      acc[sub] = wmma_bf16(a, bfr, acc[sub]);
    }
    __syncthreads();
  }

  // epilogue: bias (+residual) (+relu); f32 / bf16 / bf16^T store
#pragma unroll
  for (int sub = 0; sub < 4; ++sub)
#pragma unroll
    for (int r = 0; r < 8; ++r) {
      int row = m0 + wrow * 16 + r + half * 8;
      int cN  = n0 + wcol * 64 + sub * 16 + col;
      float v = acc[sub][r] + bias[cN];
      if (RES) v += Rsd[(size_t)row * N + cN];
      if (RELU) v = fmaxf(v, 0.f);
      if (OM == 0)      ((float*)Cv)[(size_t)row * N + cN] = v;
      else if (OM == 1) ((__bf16*)Cv)[(size_t)row * N + cN] = f2bf(v);
      else              ((__bf16*)Cv)[(size_t)cN * M + row] = f2bf(v);  // C^T
    }
}

// =====================================================================
// Flash attention (wave32 WMMA). Block = 4 waves; each wave = 16 q rows.
// Q/K bf16 row-major, V bf16 transposed [d][b*S+key] -> every fragment
// (Q,K,V) is two aligned 16B vector loads; zero converts in the hot loop.
// 1/sqrt(HD) folded into the mask select. permlane16 softmax reductions.
// =====================================================================
__global__ __launch_bounds__(128) void attn_kernel(
    const __bf16* __restrict__ Q, const __bf16* __restrict__ K,
    const __bf16* __restrict__ Vt, float* __restrict__ O,
    const int* __restrict__ ktok, int causal) {
  const int b = blockIdx.z, h = blockIdx.y;
  const int lane = threadIdx.x & 31;
  const int wave = threadIdx.x >> 5;
  const int q0 = blockIdx.x * 64 + wave * 16;
  const int col = lane & 15;
  const int half = lane >> 4;
  const float scale = 0.125f;                   // 1/sqrt(HD)
  const size_t MS = (size_t)Bx * Sx;            // V^T row stride

  const __bf16* Qb = Q + (size_t)b * Sx * DIMx + h * HDx;
  const __bf16* Kb = K + (size_t)b * Sx * DIMx + h * HDx;
  const __bf16* Vb = Vt + (size_t)(h * HDx) * MS + (size_t)b * Sx;  // [d][key]

  // Q fragments: 16x64 -> two 16x32 A-frags (raw bf16 vector loads)
  v16bf qa[2];
  {
    const __bf16* qrow = Qb + (size_t)(q0 + col) * DIMx;
#pragma unroll
    for (int f = 0; f < 2; ++f) qa[f] = load_afrag(qrow + f * 32, half * 16);
  }

  float m_run[8], l_run[8];
  v8f o_acc[4];
  {
    v8f z = {0.f, 0.f, 0.f, 0.f, 0.f, 0.f, 0.f, 0.f};
#pragma unroll
    for (int tT = 0; tT < 4; ++tT) o_acc[tT] = z;
#pragma unroll
    for (int r = 0; r < 8; ++r) { m_run[r] = -INFINITY; l_run[r] = 0.f; }
  }

  __shared__ __bf16 pbuf[4][16][32];   // per-wave P staging

  int kt_end = Sx;
  if (causal) {
    int qmax = q0 + 15;
    kt_end = ((qmax + 32) >> 5) << 5;
    if (kt_end > Sx) kt_end = Sx;
  }

  for (int kt = 0; kt < kt_end; kt += 32) {
    // prefetch next K tile rows and V^T columns (global_prefetch_b8)
    if (kt + 32 < kt_end) {
      __builtin_prefetch(&Kb[(size_t)(kt + 32 + lane) * DIMx], 0, 0);
      __builtin_prefetch(&Vb[(size_t)(lane & (HDx - 1)) * MS + kt + 32], 0, 0);
    }
    // ---- scores: 16 q rows x 32 keys, via 4 WMMAs ----
    v8f s0 = {0.f, 0.f, 0.f, 0.f, 0.f, 0.f, 0.f, 0.f};
    v8f s1 = s0;
#pragma unroll
    for (int f = 0; f < 2; ++f) {
      v16bf k0f = load_bfrag(Kb + (size_t)(kt + col) * DIMx + f * 32 + half * 16);
      v16bf k1f = load_bfrag(Kb + (size_t)(kt + 16 + col) * DIMx + f * 32 + half * 16);
      s0 = wmma_bf16(qa[f], k0f, s0);
      s1 = wmma_bf16(qa[f], k1f, s1);
    }

    // ---- scale + masking (padding + optional causal), -1e9 semantics ----
    int key0 = kt + col, key1 = kt + 16 + col;
    bool pv0 = (ktok[b * Sx + key0] != 0);
    bool pv1 = (ktok[b * Sx + key1] != 0);
#pragma unroll
    for (int r = 0; r < 8; ++r) {
      int qrow = q0 + r + half * 8;
      bool ok0 = pv0 && (!causal || key0 <= qrow);
      bool ok1 = pv1 && (!causal || key1 <= qrow);
      s0[r] = ok0 ? s0[r] * scale : NEGx;
      s1[r] = ok1 ? s1[r] * scale : NEGx;
    }

    // ---- online softmax: permlane16 butterflies, no DS ----
    float alpha[8];
#pragma unroll
    for (int r = 0; r < 8; ++r) {
      float mt = redmax16(fmaxf(s0[r], s1[r]));
      float m_new = fmaxf(m_run[r], mt);
      float p0 = __expf(s0[r] - m_new);
      float p1 = __expf(s1[r] - m_new);
      s0[r] = p0; s1[r] = p1;
      float rs = redsum16(p0 + p1);
      alpha[r] = __expf(m_run[r] - m_new);
      l_run[r] = l_run[r] * alpha[r] + rs;
      m_run[r] = m_new;
    }
#pragma unroll
    for (int tT = 0; tT < 4; ++tT)
#pragma unroll
      for (int r = 0; r < 8; ++r) o_acc[tT][r] *= alpha[r];

    // ---- P (C-layout) -> A-fragment via per-wave LDS bounce ----
#pragma unroll
    for (int r = 0; r < 8; ++r) {
      pbuf[wave][r + half * 8][col]      = f2bf(s0[r]);
      pbuf[wave][r + half * 8][16 + col] = f2bf(s1[r]);
    }
    v16bf pa;
    {
      int kb = half * 16;
#pragma unroll
      for (int i = 0; i < 8; ++i) {
        pa[i]     = pbuf[wave][col][kb + i];
        pa[8 + i] = pbuf[wave][col][8 + kb + i];
      }
    }

    // ---- O += P(16x32) x V(32x64): V^T rows are contiguous in key ----
#pragma unroll
    for (int tT = 0; tT < 4; ++tT) {
      v16bf vbf = load_bfrag(Vb + (size_t)(tT * 16 + col) * MS + kt + half * 16);
      o_acc[tT] = wmma_bf16(pa, vbf, o_acc[tT]);
    }
  }

  float* Ob = O + (size_t)b * Sx * DIMx + h * HDx;
#pragma unroll
  for (int tT = 0; tT < 4; ++tT)
#pragma unroll
    for (int r = 0; r < 8; ++r) {
      int qrow = q0 + r + half * 8;
      Ob[(size_t)qrow * DIMx + tT * 16 + col] = o_acc[tT][r] / l_run[r];
    }
}

// =====================================================================
// Embedding + sinusoidal PE (faithful to reference's 2*i/dim exponent)
// =====================================================================
__global__ __launch_bounds__(256) void embed_kernel(
    const int* __restrict__ tok, const float* __restrict__ emb,
    float* __restrict__ out) {
  int bs = blockIdx.x;
  int s = bs % Sx;
  int tk = tok[bs];
  const float sq = 22.62741699796952f;   // sqrt(512)
#pragma unroll
  for (int i = 0; i < 2; ++i) {
    int d = threadIdx.x + i * 256;
    float ang = (float)s * __powf(10000.0f, -2.0f * (float)d / (float)DIMx);
    float pe = (d & 1) ? __cosf(ang) : __sinf(ang);
    out[(size_t)bs * DIMx + d] = emb[(size_t)tk * DIMx + d] * sq + pe;
  }
}

// =====================================================================
// LayerNorm / RMSNorm: one row per 128-thread block (DIM = 512)
// =====================================================================
__global__ __launch_bounds__(128) void ln_kernel(
    const float* __restrict__ x, const float* __restrict__ g,
    const float* __restrict__ bta, float* __restrict__ out) {
  const int row = blockIdx.x;
  const float* xr = x + (size_t)row * DIMx;
  const int lane = threadIdx.x & 31, w = threadIdx.x >> 5;
  float v[4], s = 0.f;
#pragma unroll
  for (int i = 0; i < 4; ++i) { v[i] = xr[threadIdx.x + i * 128]; s += v[i]; }
#pragma unroll
  for (int m = 16; m >= 1; m >>= 1) s += __shfl_xor(s, m, 32);
  __shared__ float red[4], red2[4];
  if (lane == 0) red[w] = s;
  __syncthreads();
  s = red[0] + red[1] + red[2] + red[3];
  float mu = s / (float)DIMx;
  float q = 0.f;
#pragma unroll
  for (int i = 0; i < 4; ++i) { float d = v[i] - mu; q += d * d; }
#pragma unroll
  for (int m = 16; m >= 1; m >>= 1) q += __shfl_xor(q, m, 32);
  if (lane == 0) red2[w] = q;
  __syncthreads();
  q = red2[0] + red2[1] + red2[2] + red2[3];
  float rinv = rsqrtf(q / (float)DIMx + EPSx);
#pragma unroll
  for (int i = 0; i < 4; ++i) {
    int d = threadIdx.x + i * 128;
    out[(size_t)row * DIMx + d] = (v[i] - mu) * rinv * g[d] + bta[d];
  }
}

__global__ __launch_bounds__(128) void rms_kernel(
    const float* __restrict__ x, const float* __restrict__ w,
    float* __restrict__ out) {
  const int row = blockIdx.x;
  const float* xr = x + (size_t)row * DIMx;
  const int lane = threadIdx.x & 31, wv = threadIdx.x >> 5;
  float v[4], q = 0.f;
#pragma unroll
  for (int i = 0; i < 4; ++i) { v[i] = xr[threadIdx.x + i * 128]; q += v[i] * v[i]; }
#pragma unroll
  for (int m = 16; m >= 1; m >>= 1) q += __shfl_xor(q, m, 32);
  __shared__ float red[4];
  if (lane == 0) red[wv] = q;
  __syncthreads();
  q = red[0] + red[1] + red[2] + red[3];
  float rinv = rsqrtf(q / (float)DIMx + EPSx);
#pragma unroll
  for (int i = 0; i < 4; ++i) {
    int d = threadIdx.x + i * 128;
    out[(size_t)row * DIMx + d] = v[i] * rinv * w[d];
  }
}

// =====================================================================
// log_softmax over V=32000, in place; one row per 256-thread block
// =====================================================================
__global__ __launch_bounds__(256) void logsoftmax_kernel(float* __restrict__ p0) {
  int row = blockIdx.x;
  float* p = p0 + (size_t)row * Vx;
  const int lane = threadIdx.x & 31, w = threadIdx.x >> 5;
  __shared__ float red[8], red2[8];
  float mx = -INFINITY;
  for (int i = threadIdx.x; i < Vx; i += 256) mx = fmaxf(mx, p[i]);
#pragma unroll
  for (int m = 16; m >= 1; m >>= 1) mx = fmaxf(mx, __shfl_xor(mx, m, 32));
  if (lane == 0) red[w] = mx;
  __syncthreads();
  mx = red[0];
#pragma unroll
  for (int i = 1; i < 8; ++i) mx = fmaxf(mx, red[i]);
  float s = 0.f;
  for (int i = threadIdx.x; i < Vx; i += 256) s += __expf(p[i] - mx);
#pragma unroll
  for (int m = 16; m >= 1; m >>= 1) s += __shfl_xor(s, m, 32);
  if (lane == 0) red2[w] = s;
  __syncthreads();
  s = 0.f;
#pragma unroll
  for (int i = 0; i < 8; ++i) s += red2[i];
  float lse = mx + __logf(s);
  for (int i = threadIdx.x; i < Vx; i += 256) p[i] = p[i] - lse;
}

// =====================================================================
// Host orchestration
// =====================================================================
extern "C" void kernel_launch(void* const* d_in, const int* in_sizes, int n_in,
                              void* d_out, int out_size, void* d_ws, size_t ws_size,
                              hipStream_t stream) {
  (void)in_sizes; (void)n_in; (void)out_size; (void)ws_size;
  const int* src = (const int*)d_in[0];
  const int* tgt = (const int*)d_in[1];
  auto F = [&](int i) { return (const float*)d_in[i]; };

  float* ws = (float*)d_ws;
  const size_t BSD = (size_t)Bx * Sx * DIMx;
  float*  x   = ws; ws += BSD;
  float*  xn  = ws; ws += BSD;
  float*  att = ws; ws += BSD;
  float*  eo  = ws; ws += BSD;
  float*  hb  = ws; ws += (size_t)Bx * Sx * FFx;
  __bf16* qb  = (__bf16*)ws; ws += BSD / 2;   // bf16 Q/K row-major
  __bf16* kb  = (__bf16*)ws; ws += BSD / 2;
  __bf16* vb  = (__bf16*)ws; ws += BSD / 2;   // bf16 V transposed [d][row]

  auto gemm = [&](const float* A, const float* W, const float* bias,
                  float* C, int M, int N, int K) {
    gemm_kernel<false, false, 0><<<dim3(N / 128, M / 64), 256, 0, stream>>>(
        A, W, bias, nullptr, C, M, N, K);
  };
  auto gemm_bf = [&](const float* A, const float* W, const float* bias,
                     __bf16* C, int M, int N, int K) {
    gemm_kernel<false, false, 1><<<dim3(N / 128, M / 64), 256, 0, stream>>>(
        A, W, bias, nullptr, C, M, N, K);
  };
  auto gemm_bT = [&](const float* A, const float* W, const float* bias,
                     __bf16* C, int M, int N, int K) {
    gemm_kernel<false, false, 2><<<dim3(N / 128, M / 64), 256, 0, stream>>>(
        A, W, bias, nullptr, C, M, N, K);
  };
  auto gemm_relu = [&](const float* A, const float* W, const float* bias,
                       float* C, int M, int N, int K) {
    gemm_kernel<true, false, 0><<<dim3(N / 128, M / 64), 256, 0, stream>>>(
        A, W, bias, nullptr, C, M, N, K);
  };
  auto gemm_res = [&](const float* A, const float* W, const float* bias,
                      const float* R, float* C, int M, int N, int K) {
    gemm_kernel<false, true, 0><<<dim3(N / 128, M / 64), 256, 0, stream>>>(
        A, W, bias, R, C, M, N, K);
  };
  auto ln = [&](const float* in, const float* g, const float* b, float* out) {
    ln_kernel<<<Mx, 128, 0, stream>>>(in, g, b, out);
  };
  auto attn = [&](const int* toks, int causal) {
    attn_kernel<<<dim3(Sx / 64, Hx, Bx), 128, 0, stream>>>(qb, kb, vb, att, toks, causal);
  };

  // ================= encoder =================
  embed_kernel<<<Bx * Sx, 256, 0, stream>>>(src, F(2), x);
  for (int l = 0; l < Lx; ++l) {
    int p = 4 + 16 * l;
    ln(x, F(p + 12), F(p + 13), xn);                          // n1
    gemm_bf(xn, F(p + 0), F(p + 1), qb, Mx, DIMx, DIMx);      // q (bf16)
    gemm_bf(xn, F(p + 2), F(p + 3), kb, Mx, DIMx, DIMx);      // k (bf16)
    gemm_bT(xn, F(p + 4), F(p + 5), vb, Mx, DIMx, DIMx);      // v (bf16^T)
    attn(src, 0);
    gemm_res(att, F(p + 6), F(p + 7), x, x, Mx, DIMx, DIMx);  // o + residual
    ln(x, F(p + 14), F(p + 15), xn);                          // n2
    gemm_relu(xn, F(p + 8), F(p + 9), hb, Mx, FFx, DIMx);     // l1 + relu
    gemm_res(hb, F(p + 10), F(p + 11), x, x, Mx, DIMx, FFx);  // l2 + residual
  }
  ln(x, F(256), F(257), eo);                                  // enc_norm

  // ================= decoder (reuse x as y) =================
  embed_kernel<<<Bx * Sx, 256, 0, stream>>>(tgt, F(3), x);
  for (int l = 0; l < Lx; ++l) {
    int p = 100 + 26 * l;
    ln(x, F(p + 20), F(p + 21), xn);                          // n1
    gemm_bf(xn, F(p + 0), F(p + 1), qb, Mx, DIMx, DIMx);
    gemm_bf(xn, F(p + 2), F(p + 3), kb, Mx, DIMx, DIMx);
    gemm_bT(xn, F(p + 4), F(p + 5), vb, Mx, DIMx, DIMx);
    attn(tgt, 1);
    gemm_res(att, F(p + 6), F(p + 7), x, x, Mx, DIMx, DIMx);
    ln(x, F(p + 22), F(p + 23), xn);                          // n2
    gemm_bf(xn, F(p + 8), F(p + 9), qb, Mx, DIMx, DIMx);
    gemm_bf(eo, F(p + 10), F(p + 11), kb, Mx, DIMx, DIMx);
    gemm_bT(eo, F(p + 12), F(p + 13), vb, Mx, DIMx, DIMx);
    attn(src, 0);
    gemm_res(att, F(p + 14), F(p + 15), x, x, Mx, DIMx, DIMx);
    ln(x, F(p + 24), F(p + 25), xn);                          // n3
    gemm_relu(xn, F(p + 16), F(p + 17), hb, Mx, FFx, DIMx);
    gemm_res(hb, F(p + 18), F(p + 19), x, x, Mx, DIMx, FFx);
  }
  rms_kernel<<<Mx, 128, 0, stream>>>(x, F(258), x);           // dec_norm (RMS)

  gemm(x, F(259), F(260), (float*)d_out, Mx, Vx, DIMx);
  logsoftmax_kernel<<<Mx, 256, 0, stream>>>((float*)d_out);
}
